// HungarianMatcher_25735444038169
// MI455X (gfx1250) — matrix operand order repaired and checked by reading, not compiled, not run on hardware
//
#include <hip/hip_runtime.h>

typedef __attribute__((ext_vector_type(16))) _Float16 v16h;
typedef __attribute__((ext_vector_type(8)))  _Float16 v8h;
typedef __attribute__((ext_vector_type(4)))  _Float16 v4h;
typedef __attribute__((ext_vector_type(8)))  float    v8f;
typedef __attribute__((ext_vector_type(4)))  float    v4f;

#define NQ 19200   // 64 * 300 queries
#define NT 1920    // targets
#define NL 256     // logit dim

// ---------------------------------------------------------------------------
// Kernel 1: row softmax (f32 -> f16), one wave32 per row of 256 logits.
// Each lane owns 8 consecutive floats; cross-lane reductions via shfl_xor.
// ---------------------------------------------------------------------------
__global__ __launch_bounds__(256)
void softmax_rows_f16(const float* __restrict__ logits, _Float16* __restrict__ prob) {
    const int row  = (blockIdx.x * blockDim.x + threadIdx.x) >> 5;
    const int lane = threadIdx.x & 31;
    const float* rp = logits + (size_t)row * NL + lane * 8;
    v4f a = *(const v4f*)(rp);
    v4f b = *(const v4f*)(rp + 4);
    float va[8] = {a.x, a.y, a.z, a.w, b.x, b.y, b.z, b.w};

    float m = va[0];
#pragma unroll
    for (int i = 1; i < 8; ++i) m = fmaxf(m, va[i]);
#pragma unroll
    for (int off = 16; off >= 1; off >>= 1) m = fmaxf(m, __shfl_xor(m, off, 32));

    float s = 0.f;
#pragma unroll
    for (int i = 0; i < 8; ++i) { va[i] = __expf(va[i] - m); s += va[i]; }
#pragma unroll
    for (int off = 16; off >= 1; off >>= 1) s += __shfl_xor(s, off, 32);

    const float inv = 1.0f / s;
    v8h ph;
#pragma unroll
    for (int i = 0; i < 8; ++i) ph[i] = (_Float16)(va[i] * inv);
    *(v8h*)(prob + (size_t)row * NL + lane * 8) = ph;
}

// ---------------------------------------------------------------------------
// Kernel 2: positive_map f32 -> f16, 4 elements per thread (exact cover).
// ---------------------------------------------------------------------------
__global__ __launch_bounds__(256)
void cvt_f32_to_f16x4(const float* __restrict__ src, _Float16* __restrict__ dst) {
    const int i = blockIdx.x * blockDim.x + threadIdx.x;
    v4f s = *(const v4f*)(src + (size_t)i * 4);
    v4h d;
    d[0] = (_Float16)s.x; d[1] = (_Float16)s.y;
    d[2] = (_Float16)s.z; d[3] = (_Float16)s.w;
    *(v4h*)(dst + (size_t)i * 4) = d;
}

// ---------------------------------------------------------------------------
// Kernel 3: fused cost matrix.
//   Block: 256 threads = 8 waves. Wave w -> rows [m_base, m_base+16),
//   strip of 64 targets. Per wave: 4 f32 accumulators (16x16 each),
//   K-loop 256 in steps of 32 -> 32x v_wmma_f32_16x16x32_f16.
//   Epilogue fuses L1-bbox and GIoU directly on the accumulator layout:
//   acc[nt][v] is element (M = 8*(lane/16)+v, N = lane%16) of tile nt.
// ---------------------------------------------------------------------------
__global__ __launch_bounds__(256)
void cost_matrix_wmma(const _Float16* __restrict__ Ah,   // [NQ, NL] f16 probs
                      const _Float16* __restrict__ Bh,   // [NT, NL] f16 pos map
                      const float*    __restrict__ pred_boxes, // [NQ,4] cxcywh
                      const float*    __restrict__ tgt_boxes,  // [NT,4] cxcywh
                      float*          __restrict__ C)          // [NQ, NT]
{
    const int lane = threadIdx.x & 31;
    const int wave = threadIdx.x >> 5;
    const int r = lane & 15;   // 0..15
    const int g = lane >> 4;   // 0..1

    const int m_base = blockIdx.y * 128 + wave * 16;
    const int n_base = blockIdx.x * 64;

    v8f acc[4] = {};

    const _Float16* aRow = Ah + (size_t)(m_base + r) * NL;

#pragma unroll
    for (int k0 = 0; k0 < NL; k0 += 32) {
        if (k0 + 32 < NL) __builtin_prefetch(aRow + k0 + 32, 0, 0);
        // A fragment (16x32 f16): lane(r,g) holds 16 contiguous K = 16g..16g+15
        v8h a0 = *(const v8h*)(aRow + k0 + 16 * g);
        v8h a1 = *(const v8h*)(aRow + k0 + 8 + 16 * g);
        v16h afrag = __builtin_shufflevector(a0, a1,
                       0,1,2,3,4,5,6,7,8,9,10,11,12,13,14,15);
#pragma unroll
        for (int nt = 0; nt < 4; ++nt) {
            // B fragment (32x16 f16): column N=r is row t of positive_map;
            // lane(r,g) holds 16 contiguous halfs at K = k0 + 16*g.
            const _Float16* bp =
                Bh + (size_t)(n_base + nt * 16 + r) * NL + k0 + 16 * g;
            v16h bfrag = *(const v16h*)bp;
            acc[nt] = __builtin_amdgcn_wmma_f32_16x16x32_f16(
                /*neg_a=*/false, afrag, /*neg_b=*/false, bfrag,
                /*c_mod=*/(short)0, acc[nt], /*reuse_a=*/false, /*reuse_b=*/false);
        }
    }

    // --- epilogue: L1 + GIoU fused on accumulator layout ---
    v4f pb[8];
#pragma unroll
    for (int v = 0; v < 8; ++v) {
        const int q = m_base + 8 * g + v;
        pb[v] = *(const v4f*)(pred_boxes + (size_t)q * 4);
    }

#pragma unroll
    for (int nt = 0; nt < 4; ++nt) {
        const int t = n_base + nt * 16 + r;
        v4f tb = *(const v4f*)(tgt_boxes + (size_t)t * 4);
        const float tx0 = tb.x - 0.5f * tb.z, ty0 = tb.y - 0.5f * tb.w;
        const float tx1 = tb.x + 0.5f * tb.z, ty1 = tb.y + 0.5f * tb.w;
        const float tarea = (tx1 - tx0) * (ty1 - ty0);
#pragma unroll
        for (int v = 0; v < 8; ++v) {
            const int q = m_base + 8 * g + v;
            const v4f p = pb[v];
            const float l1 = fabsf(p.x - tb.x) + fabsf(p.y - tb.y) +
                             fabsf(p.z - tb.z) + fabsf(p.w - tb.w);
            const float px0 = p.x - 0.5f * p.z, py0 = p.y - 0.5f * p.w;
            const float px1 = p.x + 0.5f * p.z, py1 = p.y + 0.5f * p.w;
            const float parea = (px1 - px0) * (py1 - py0);
            const float iw = fminf(px1, tx1) - fmaxf(px0, tx0);
            const float ih = fminf(py1, ty1) - fmaxf(py0, ty0);
            const float inter = fmaxf(iw, 0.f) * fmaxf(ih, 0.f);
            const float uni   = parea + tarea - inter;
            const float iou   = inter / uni;
            const float ew = fmaxf(px1, tx1) - fminf(px0, tx0);
            const float eh = fmaxf(py1, ty1) - fminf(py0, ty0);
            const float earea = fmaxf(ew, 0.f) * fmaxf(eh, 0.f);
            const float giou  = iou - (earea - uni) / earea;
            // C = 1*l1 + 1*(-dot) + 1*(-giou)
            C[(size_t)q * NT + t] = l1 - acc[nt][v] - giou;
        }
    }
}

// ---------------------------------------------------------------------------
extern "C" void kernel_launch(void* const* d_in, const int* in_sizes, int n_in,
                              void* d_out, int out_size, void* d_ws, size_t ws_size,
                              hipStream_t stream) {
    const float* pred_logits  = (const float*)d_in[0]; // [64,300,256]
    const float* pred_boxes   = (const float*)d_in[1]; // [64,300,4]
    const float* tgt_boxes    = (const float*)d_in[2]; // [1920,4]
    const float* positive_map = (const float*)d_in[3]; // [1920,256]
    float* out = (float*)d_out;                        // [64,300,1920]

    // workspace: f16 probs (9.83 MB) then f16 positive_map (0.98 MB)
    _Float16* prob_h = (_Float16*)d_ws;
    _Float16* pm_h   = (_Float16*)((char*)d_ws + (size_t)NQ * NL * 2);

    // 1) softmax: 19200 rows, 8 waves (rows) per 256-thread block
    softmax_rows_f16<<<NQ / 8, 256, 0, stream>>>(pred_logits, prob_h);

    // 2) positive_map f32 -> f16: 1920*256/4 threads, exact cover
    cvt_f32_to_f16x4<<<(NT * NL / 4) / 256, 256, 0, stream>>>(positive_map, pm_h);

    // 3) fused WMMA cost matrix: block covers 128(M) x 64(N)
    dim3 grid(NT / 64, NQ / 128);
    cost_matrix_wmma<<<grid, 256, 0, stream>>>(prob_h, pm_h, pred_boxes,
                                               tgt_boxes, out);
}
